// Fusion_KAN_50654844289355
// MI455X (gfx1250) — compile-verified, matrix operand order checked by reading, not run on hardware
//
#include <hip/hip_runtime.h>
#include <hip/hip_bf16.h>
#include <math.h>

typedef __attribute__((ext_vector_type(16))) _Float16 v16h;
typedef __attribute__((ext_vector_type(8)))  _Float16 h8;
typedef __attribute__((ext_vector_type(4)))  _Float16 h4;
typedef __attribute__((ext_vector_type(8)))  float    v8f;

#define CDIM   512
#define KCAT   4608   // 512 (silu) + 512*8 (spline bases)
#define NROWS  30720  // 32*(512+256+128+64)
#define BC     16384  // B*C rows for mixer GEMMs

__device__ __forceinline__ float gelu_exact(float x) {
    return 0.5f * x * (1.0f + erff(x * 0.70710678118654752f));
}
__device__ __forceinline__ float silu_f(float x) {
    return x / (1.0f + expf(-x));
}

// ---- fragment loader: 16x32 f16 tile from LDS (row stride = 40 halves)
// ISA 16-bit A/B layout: lanes 0-15 -> K {0..7,16..23}; lanes 16-31 -> K {8..15,24..31}
__device__ __forceinline__ v16h load_frag(const _Float16* p, int lane) {
    int r  = lane & 15;
    int kk = (lane >> 4) << 3;            // 0 or 8
    h8 lo = *(const h8*)(p + r * 40 + kk);
    h8 hi = *(const h8*)(p + r * 40 + kk + 16);
    v16h f;
#pragma unroll
    for (int j = 0; j < 8; ++j) { f[j] = lo[j]; f[j + 8] = hi[j]; }
    return f;
}

// =====================================================================
// Mixer GEMM: Out(N,M) = epi( A(N,K) @ W(M,K)^T + bias [+ Res] )
// fp32 inputs, f16 WMMA math, fp32 out. BN=128, BM=64, BK=32, 256 thr.
// K, M compile-time -> immediate-offset epilogue stores.
// =====================================================================
template <int ACT, int HAS_RES, int K, int M>
__global__ __launch_bounds__(256) void mix_gemm(const float* __restrict__ A,
                                                const float* __restrict__ W,
                                                const float* __restrict__ bias,
                                                const float* __restrict__ Res,
                                                float* __restrict__ Out) {
    __shared__ _Float16 As[128 * 40];
    __shared__ _Float16 Ws[64 * 40];

    const int tid  = threadIdx.x;
    const int lane = tid & 31;
    const int wave = tid >> 5;          // 0..7
    const int wr   = wave & 3;          // row group (32 rows)
    const int wc   = wave >> 2;         // col group (32 cols)
    const int bx   = blockIdx.x;        // M / 64
    const int by   = blockIdx.y;        // N / 128

    v8f acc[2][2];
    v8f zero = {};
#pragma unroll
    for (int i = 0; i < 2; ++i)
#pragma unroll
        for (int j = 0; j < 2; ++j) acc[i][j] = zero;

    for (int kt = 0; kt < K; kt += 32) {
        // A tile: 128x32 fp32 -> f16 LDS
#pragma unroll
        for (int it = 0; it < 4; ++it) {
            int idx = tid + it * 256;            // 0..1023 float4 chunks
            int row = idx >> 3;
            int c4  = (idx & 7) << 2;
            float4 v = *(const float4*)(A + (size_t)(by * 128 + row) * K + kt + c4);
            h4 hv = { (_Float16)v.x, (_Float16)v.y, (_Float16)v.z, (_Float16)v.w };
            *(h4*)&As[row * 40 + c4] = hv;
        }
        // W tile: 64x32 fp32 -> f16 LDS
#pragma unroll
        for (int it = 0; it < 2; ++it) {
            int idx = tid + it * 256;            // 0..511
            int row = idx >> 3;
            int c4  = (idx & 7) << 2;
            float4 v = *(const float4*)(W + (size_t)(bx * 64 + row) * K + kt + c4);
            h4 hv = { (_Float16)v.x, (_Float16)v.y, (_Float16)v.z, (_Float16)v.w };
            *(h4*)&Ws[row * 40 + c4] = hv;
        }
        __syncthreads();

        v16h bfr[2], afr[2];
#pragma unroll
        for (int j = 0; j < 2; ++j) bfr[j] = load_frag(&Ws[(wc * 32 + j * 16) * 40], lane);
#pragma unroll
        for (int i = 0; i < 2; ++i) afr[i] = load_frag(&As[(wr * 32 + i * 16) * 40], lane);
#pragma unroll
        for (int i = 0; i < 2; ++i)
#pragma unroll
            for (int j = 0; j < 2; ++j)
                acc[i][j] = __builtin_amdgcn_wmma_f32_16x16x32_f16(
                    false, afr[i], false, bfr[j], (short)0, acc[i][j], false, false);
        __syncthreads();
    }

    // Epilogue. D layout: VGPR r, lanes 0-15 -> M=r,N=lane; lanes 16-31 -> M=r+8
    const int lhi = lane >> 4;
    const int ln  = lane & 15;
#pragma unroll
    for (int j = 0; j < 2; ++j) {
        int col = bx * 64 + wc * 32 + j * 16 + ln;
        float bv = bias[col];
#pragma unroll
        for (int i = 0; i < 2; ++i) {
            int row0 = by * 128 + wr * 32 + i * 16 + lhi * 8;
            float* __restrict__ op = Out + (size_t)row0 * M + col;
            const float* __restrict__ rp = HAS_RES ? (Res + (size_t)row0 * M + col) : nullptr;
#pragma unroll
            for (int r = 0; r < 8; ++r) {
                float v = acc[i][j][r] + bv;
                if (ACT == 1) v = gelu_exact(v);
                if (HAS_RES)  v += rp[r * M];
                op[r * M] = v;   // immediate offsets (M compile-time)
            }
        }
    }
}

// =====================================================================
// KAN GEMM: Out(N,512) = Acat(N,4608)f16 @ Wcat(512,4608)^T f16
// BN=128, BM=128, BK=32, 256 threads, 8 waves (4 row x 2 col groups),
// each wave: 2 row tiles x 4 col tiles => 8 WMMA per K-step.
// Double-buffered LDS + register prefetch: overlap global loads w/ WMMA.
// =====================================================================
__global__ __launch_bounds__(256) void kan_gemm(const _Float16* __restrict__ A,
                                                const _Float16* __restrict__ W,
                                                float* __restrict__ Out) {
    constexpr int K = KCAT;
    constexpr int M = CDIM;
    __shared__ _Float16 As[2][128 * 40];
    __shared__ _Float16 Ws[2][128 * 40];

    const int tid  = threadIdx.x;
    const int lane = tid & 31;
    const int wave = tid >> 5;
    const int wr   = wave & 3;    // 32 rows
    const int wc   = wave >> 2;   // 64 cols
    const int bx   = blockIdx.x;  // M/128
    const int by   = blockIdx.y;  // N/128

    const _Float16* __restrict__ Ap = A + (size_t)(by * 128) * K;
    const _Float16* __restrict__ Wp = W + (size_t)(bx * 128) * K;

    // per-thread staging coords: 512 x 16B chunks per tile -> 2 per thread
    const int r0 = tid >> 2;                 // row for chunk 0 (0..63)
    const int c8 = (tid & 3) << 3;           // k offset within tile

    h8 ra[2], rw[2];
    auto gload = [&](int kt) {
#pragma unroll
        for (int it = 0; it < 2; ++it) {
            int row = r0 + it * 64;
            ra[it] = *(const h8*)(Ap + (size_t)row * K + kt + c8);
            rw[it] = *(const h8*)(Wp + (size_t)row * K + kt + c8);
        }
    };
    auto sstore = [&](int buf) {
#pragma unroll
        for (int it = 0; it < 2; ++it) {
            int row = r0 + it * 64;
            *(h8*)&As[buf][row * 40 + c8] = ra[it];
            *(h8*)&Ws[buf][row * 40 + c8] = rw[it];
        }
    };

    v8f acc[2][4];
    v8f zero = {};
#pragma unroll
    for (int i = 0; i < 2; ++i)
#pragma unroll
        for (int j = 0; j < 4; ++j) acc[i][j] = zero;

    gload(0);
    sstore(0);
    __syncthreads();

    constexpr int NT = K / 32;   // 144
    for (int it = 0; it < NT; ++it) {
        const int curb = it & 1;
        const bool more = (it + 1) < NT;
        if (more) gload((it + 1) * 32);

        v16h bfr[4], afr[2];
#pragma unroll
        for (int j = 0; j < 4; ++j) bfr[j] = load_frag(&Ws[curb][(wc * 64 + j * 16) * 40], lane);
#pragma unroll
        for (int i = 0; i < 2; ++i) afr[i] = load_frag(&As[curb][(wr * 32 + i * 16) * 40], lane);
#pragma unroll
        for (int i = 0; i < 2; ++i)
#pragma unroll
            for (int j = 0; j < 4; ++j)
                acc[i][j] = __builtin_amdgcn_wmma_f32_16x16x32_f16(
                    false, afr[i], false, bfr[j], (short)0, acc[i][j], false, false);

        if (more) sstore(curb ^ 1);
        __syncthreads();
    }

    const int lhi = lane >> 4;
    const int ln  = lane & 15;
#pragma unroll
    for (int j = 0; j < 4; ++j) {
        int col = bx * 128 + wc * 64 + j * 16 + ln;
#pragma unroll
        for (int i = 0; i < 2; ++i) {
            int row0 = by * 128 + wr * 32 + i * 16 + lhi * 8;
            float* __restrict__ op = Out + (size_t)row0 * M + col;
#pragma unroll
            for (int r = 0; r < 8; ++r) op[r * M] = acc[i][j][r];
        }
    }
}

// =====================================================================
// season(B,C,T) + trend(B,C,T) -> O rows (rowOff + b*T + t), cols C
// =====================================================================
__global__ __launch_bounds__(256) void transpose_add(const float* __restrict__ S,
                                                     const float* __restrict__ Tr,
                                                     float* __restrict__ O,
                                                     int T, int rowOff) {
    __shared__ float tile[32][33];
    int b  = blockIdx.z;
    int t0 = blockIdx.x * 32;
    int c0 = blockIdx.y * 32;
#pragma unroll
    for (int i = 0; i < 4; ++i) {
        int c = c0 + threadIdx.y + i * 8;
        int t = t0 + threadIdx.x;
        size_t idx = ((size_t)b * CDIM + c) * T + t;
        tile[threadIdx.y + i * 8][threadIdx.x] = S[idx] + Tr[idx];
    }
    __syncthreads();
#pragma unroll
    for (int i = 0; i < 4; ++i) {
        int t = t0 + threadIdx.y + i * 8;
        int c = c0 + threadIdx.x;
        O[((size_t)(rowOff + b * T + t)) * CDIM + c] = tile[threadIdx.x][threadIdx.y + i * 8];
    }
}

// =====================================================================
// Wcat[l][o][0:512]=bw ; Wcat[l][o][512+i*8+k]=sw*sc  (f16)
// =====================================================================
__global__ __launch_bounds__(256) void prep_weights(const float* __restrict__ bw,
                                                    const float* __restrict__ sw,
                                                    const float* __restrict__ sc,
                                                    _Float16* __restrict__ Wcat) {
    size_t idx = (size_t)blockIdx.x * 256 + threadIdx.x;
    if (idx >= (size_t)3 * CDIM * KCAT) return;
    size_t lo = idx / KCAT;      // l*512 + o
    int    j  = (int)(idx % KCAT);
    float v;
    if (j < CDIM) {
        v = bw[lo * CDIM + j];
    } else {
        int ij = j - CDIM;
        int i  = ij >> 3;
        int k  = ij & 7;
        v = sw[(lo * CDIM + i) * 8 + k] * sc[lo * CDIM + i];
    }
    Wcat[idx] = (_Float16)v;
}

// =====================================================================
// Acat[n][0:512] = silu(x); Acat[n][512+i*8+k] = cubic B-spline basis
// uniform grid: g_j = (j-3)*0.4 - 1, j=0..11
// =====================================================================
__global__ __launch_bounds__(256) void kan_prep(const float* __restrict__ X,
                                                _Float16* __restrict__ Acat) {
    size_t idx = (size_t)blockIdx.x * 256 + threadIdx.x;
    if (idx >= (size_t)NROWS * CDIM) return;
    size_t n = idx >> 9;
    int    i = (int)(idx & 511);
    float x = X[idx];

    Acat[n * KCAT + i] = (_Float16)silu_f(x);

    const float h = 0.4f;
    float g[12];
#pragma unroll
    for (int j = 0; j < 12; ++j) g[j] = (float)(j - 3) * h - 1.0f;
    float b[11];
#pragma unroll
    for (int j = 0; j < 11; ++j) b[j] = (x >= g[j] && x < g[j + 1]) ? 1.0f : 0.0f;
#pragma unroll
    for (int k = 1; k <= 3; ++k) {
        float inv = 1.0f / ((float)k * h);
#pragma unroll
        for (int j = 0; j < 10; ++j) {
            if (j < 11 - k) {
                float left  = (x - g[j]) * inv * b[j];
                float right = (g[j + k + 1] - x) * inv * b[j + 1];
                b[j] = left + right;
            }
        }
    }
    h8 out;
#pragma unroll
    for (int k = 0; k < 8; ++k) out[k] = (_Float16)b[k];
    *(h8*)(Acat + n * KCAT + CDIM + (size_t)i * 8) = out;
}

__global__ __launch_bounds__(256) void final_add(const float* __restrict__ Xres,
                                                 const float* __restrict__ O2,
                                                 float* __restrict__ Out, int n) {
    int i = blockIdx.x * 256 + threadIdx.x;
    if (i < n) Out[i] = Xres[i] + O2[i];
}

// =====================================================================
extern "C" void kernel_launch(void* const* d_in, const int* in_sizes, int n_in,
                              void* d_out, int out_size, void* d_ws, size_t ws_size,
                              hipStream_t stream) {
    // inputs (setup_inputs order)
    const float* x[4] = { (const float*)d_in[0], (const float*)d_in[3],
                          (const float*)d_in[6], (const float*)d_in[9] };
    const float* s[4] = { (const float*)d_in[1], (const float*)d_in[4],
                          (const float*)d_in[7], (const float*)d_in[10] };
    const float* t[4] = { (const float*)d_in[2], (const float*)d_in[5],
                          (const float*)d_in[8], (const float*)d_in[11] };
    const float* dw1[3] = { (const float*)d_in[12], (const float*)d_in[16], (const float*)d_in[20] };
    const float* db1[3] = { (const float*)d_in[13], (const float*)d_in[17], (const float*)d_in[21] };
    const float* dw2[3] = { (const float*)d_in[14], (const float*)d_in[18], (const float*)d_in[22] };
    const float* db2[3] = { (const float*)d_in[15], (const float*)d_in[19], (const float*)d_in[23] };
    const float* uw1[3] = { (const float*)d_in[24], (const float*)d_in[28], (const float*)d_in[32] };
    const float* ub1[3] = { (const float*)d_in[25], (const float*)d_in[29], (const float*)d_in[33] };
    const float* uw2[3] = { (const float*)d_in[26], (const float*)d_in[30], (const float*)d_in[34] };
    const float* ub2[3] = { (const float*)d_in[27], (const float*)d_in[31], (const float*)d_in[35] };
    const float* kan_bw = (const float*)d_in[36];
    const float* kan_sw = (const float*)d_in[37];
    const float* kan_sc = (const float*)d_in[38];

    // workspace carve-up
    char*  base = (char*)d_ws;
    size_t cur  = 0;
    auto alloc = [&](size_t bytes) -> char* {
        char* p = base + cur;
        cur += (bytes + 255) & ~(size_t)255;
        return p;
    };
    float*     g1    = (float*)alloc((size_t)BC * 512 * 4);
    float*     sea1  = (float*)alloc((size_t)BC * 256 * 4);
    float*     sea2  = (float*)alloc((size_t)BC * 128 * 4);
    float*     sea3  = (float*)alloc((size_t)BC * 64 * 4);
    float*     tr2   = (float*)alloc((size_t)BC * 128 * 4);
    float*     tr1   = (float*)alloc((size_t)BC * 256 * 4);
    float*     tr0   = (float*)alloc((size_t)BC * 512 * 4);
    float*     xbuf0 = (float*)alloc((size_t)NROWS * CDIM * 4);
    float*     xbuf1 = (float*)alloc((size_t)NROWS * CDIM * 4);
    _Float16*  Acat  = (_Float16*)alloc((size_t)NROWS * KCAT * 2);
    _Float16*  Wcat  = (_Float16*)alloc((size_t)3 * CDIM * KCAT * 2);
    (void)ws_size; (void)in_sizes; (void)n_in; (void)out_size;

    const dim3 thr(256);
    auto mgrid = [](int M) { return dim3(M / 64, BC / 128); };

    // ---------------- down (season) chain ----------------
    mix_gemm<1, 0, 512, 256><<<mgrid(256), thr, 0, stream>>>(s[0], dw1[0], db1[0], nullptr, g1);
    mix_gemm<0, 1, 256, 256><<<mgrid(256), thr, 0, stream>>>(g1, dw2[0], db2[0], s[1], sea1);
    mix_gemm<1, 0, 256, 128><<<mgrid(128), thr, 0, stream>>>(sea1, dw1[1], db1[1], nullptr, g1);
    mix_gemm<0, 1, 128, 128><<<mgrid(128), thr, 0, stream>>>(g1, dw2[1], db2[1], s[2], sea2);
    mix_gemm<1, 0, 128, 64><<<mgrid(64), thr, 0, stream>>>(sea2, dw1[2], db1[2], nullptr, g1);
    mix_gemm<0, 1, 64, 64><<<mgrid(64), thr, 0, stream>>>(g1, dw2[2], db2[2], s[3], sea3);

    // ---------------- up (trend) chain ----------------
    mix_gemm<1, 0, 64, 128><<<mgrid(128), thr, 0, stream>>>(t[3], uw1[0], ub1[0], nullptr, g1);
    mix_gemm<0, 1, 128, 128><<<mgrid(128), thr, 0, stream>>>(g1, uw2[0], ub2[0], t[2], tr2);
    mix_gemm<1, 0, 128, 256><<<mgrid(256), thr, 0, stream>>>(tr2, uw1[1], ub1[1], nullptr, g1);
    mix_gemm<0, 1, 256, 256><<<mgrid(256), thr, 0, stream>>>(g1, uw2[1], ub2[1], t[1], tr1);
    mix_gemm<1, 0, 256, 512><<<mgrid(512), thr, 0, stream>>>(tr1, uw1[2], ub1[2], nullptr, g1);
    mix_gemm<0, 1, 512, 512><<<mgrid(512), thr, 0, stream>>>(g1, uw2[2], ub2[2], t[0], tr0);

    // ---------------- o = season + trend (transposed, concatenated) ----------------
    {
        dim3 tb(32, 8);
        transpose_add<<<dim3(512 / 32, 16, 32), tb, 0, stream>>>(s[0], tr0, xbuf0, 512, 0);
        transpose_add<<<dim3(256 / 32, 16, 32), tb, 0, stream>>>(sea1, tr1, xbuf0, 256, 16384);
        transpose_add<<<dim3(128 / 32, 16, 32), tb, 0, stream>>>(sea2, tr2, xbuf0, 128, 24576);
        transpose_add<<<dim3(64 / 32, 16, 32), tb, 0, stream>>>(sea3, t[3], xbuf0, 64, 28672);
    }

    // ---------------- KAN weights ----------------
    {
        size_t n = (size_t)3 * CDIM * KCAT;
        prep_weights<<<(unsigned)((n + 255) / 256), thr, 0, stream>>>(kan_bw, kan_sw, kan_sc, Wcat);
    }

    // ---------------- KAN layers ----------------
    float* curx = xbuf0;
    float* nxtx = xbuf1;
    for (int l = 0; l < 3; ++l) {
        size_t n = (size_t)NROWS * CDIM;
        kan_prep<<<(unsigned)((n + 255) / 256), thr, 0, stream>>>(curx, Acat);
        kan_gemm<<<dim3(CDIM / 128, NROWS / 128), thr, 0, stream>>>(
            Acat, Wcat + (size_t)l * CDIM * KCAT, nxtx);
        float* tmp = curx; curx = nxtx; nxtx = tmp;
    }

    // ---------------- residual add into d_out (layouts match flat concat) ----------------
    {
        float* out = (float*)d_out;
        const int Ts[4] = { 512, 256, 128, 64 };
        size_t off = 0;
        for (int q = 0; q < 4; ++q) {
            int n = 32 * Ts[q] * CDIM;
            final_add<<<(n + 255) / 256, thr, 0, stream>>>(x[q], curx + off, out + off, n);
            off += (size_t)n;
        }
    }
}
